// RowAttentionWithPairBias_9268539425318
// MI455X (gfx1250) — matrix operand order repaired and checked by reading, not compiled, not run on hardware
//
#include <hip/hip_runtime.h>
#include <hip/hip_bf16.h>
#include <math.h>

// ---------------------------------------------------------------------------
// RowAttentionWithPairBias for MI455X (gfx1250), wave32 + WMMA bf16.
// B=1, M=128, N=256, C_IN=256, C_PAIR=128, H=8, C=32.
// ---------------------------------------------------------------------------

typedef __bf16 bf16;
typedef __attribute__((ext_vector_type(16))) __bf16 bf16x16;
typedef __attribute__((ext_vector_type(8)))  __bf16 bf16x8;
typedef __attribute__((ext_vector_type(8)))  float  f32x8;
typedef __attribute__((ext_vector_type(4)))  float  f32x4;

#define WMMA_BF16(A, B, C) \
  __builtin_amdgcn_wmma_f32_16x16x32_bf16(false, (A), false, (B), (short)0, (C), false, false)

// ---- WMMA fragment assembly per CDNA5 ISA VGPR layouts (wave32) -----------
// A 16x32 bf16: lane l holds row l%16; a[0..7]=K(koff..koff+7), a[8..15]=K(16+koff..23+koff), koff=8*(l>=16)
__device__ __forceinline__ bf16x16 frag_a(const bf16* __restrict__ rowptr, int lane) {
  int koff = (lane >> 4) << 3;
  union { bf16x16 v; bf16x8 h[2]; } u;
  u.h[0] = *(const bf16x8*)(rowptr + koff);
  u.h[1] = *(const bf16x8*)(rowptr + 16 + koff);
  return u.v;
}
// B 32x16 bf16 from BT[n][k] storage: lane l holds col l%16; b[0..7]=K(koff..koff+7), b[8..15]=K(16+koff..23+koff), koff=8*(l>=16)
__device__ __forceinline__ bf16x16 frag_b(const bf16* __restrict__ colptr, int lane) {
  const bf16* p = colptr + ((lane >> 4) << 3);
  union { bf16x16 v; bf16x8 h[2]; } u;
  u.h[0] = *(const bf16x8*)(p);
  u.h[1] = *(const bf16x8*)(p + 16);
  return u.v;
}
// D/C f32: d[v] -> row M = v + 8*(l>=16), col N = l%16

// ---------------------------------------------------------------------------
// K0: transpose + bf16-cast all weight matrices. WT[o][i] = W[i][o] (* 1/sqrt(C) for Wq)
// z=0..3 -> WqT,WkT,WvT,WgateT ; z=4 -> WfinalT. 5*65536 threads.
__global__ void k_prep_weights(const float* __restrict__ Wq, const float* __restrict__ Wk,
                               const float* __restrict__ Wv, const float* __restrict__ Wg,
                               const float* __restrict__ Wf, bf16* __restrict__ wT) {
  int idx = blockIdx.x * blockDim.x + threadIdx.x;
  int z = idx >> 16;
  int t = idx & 65535;
  int o = t >> 8, i = t & 255;
  const float* W;
  float scale = 1.0f;
  if (z == 0)      { W = Wq; scale = 0.17677669529663687f; }  // 1/sqrt(32)
  else if (z == 1) { W = Wk; }
  else if (z == 2) { W = Wv; }
  else if (z == 3) { W = Wg; }
  else             { W = Wf; }
  wT[z * 65536 + o * 256 + i] = (bf16)(W[i * 256 + o] * scale);
}

// ---------------------------------------------------------------------------
// K1: LayerNorm over C_IN=256, output bf16. One wave per row (8 elems/lane).
__global__ void k_ln1(const float* __restrict__ x, const float* __restrict__ g,
                      const float* __restrict__ b, bf16* __restrict__ out) {
  int lane = threadIdx.x & 31;
  int row  = blockIdx.x * (blockDim.x >> 5) + (threadIdx.x >> 5);  // 0..32767
  const float* xr = x + row * 256 + lane * 8;
  f32x4 v0 = *(const f32x4*)xr;
  f32x4 v1 = *(const f32x4*)(xr + 4);
  float s = 0.f, ss = 0.f;
#pragma unroll
  for (int j = 0; j < 4; ++j) {
    s  += v0[j] + v1[j];
    ss += v0[j] * v0[j] + v1[j] * v1[j];
  }
#pragma unroll
  for (int o = 16; o > 0; o >>= 1) { s += __shfl_xor(s, o, 32); ss += __shfl_xor(ss, o, 32); }
  float mu  = s * (1.0f / 256.0f);
  float var = ss * (1.0f / 256.0f) - mu * mu;
  float inv = rsqrtf(var + 1e-5f);
  bf16x8 o8;
#pragma unroll
  for (int j = 0; j < 4; ++j) {
    o8[j]     = (bf16)((v0[j] - mu) * inv * g[lane * 8 + j]     + b[lane * 8 + j]);
    o8[j + 4] = (bf16)((v1[j] - mu) * inv * g[lane * 8 + 4 + j] + b[lane * 8 + 4 + j]);
  }
  *(bf16x8*)(out + row * 256 + lane * 8) = o8;
}

// ---------------------------------------------------------------------------
// K2: LayerNorm over C_PAIR=128 + bias projection: bias[h][i][j] = sum_p x2n[p]*W2d[p][h]
// One wave per (i,j) row; 4 elems/lane.
__global__ void k_ln2_bias(const float* __restrict__ x2, const float* __restrict__ g,
                           const float* __restrict__ b, const float* __restrict__ W2d,
                           float* __restrict__ bias) {
  int lane = threadIdx.x & 31;
  int row  = blockIdx.x * (blockDim.x >> 5) + (threadIdx.x >> 5);  // 0..65535
  const float* xr = x2 + row * 128 + lane * 4;
  f32x4 v = *(const f32x4*)xr;
  float s = 0.f, ss = 0.f;
#pragma unroll
  for (int j = 0; j < 4; ++j) { s += v[j]; ss += v[j] * v[j]; }
#pragma unroll
  for (int o = 16; o > 0; o >>= 1) { s += __shfl_xor(s, o, 32); ss += __shfl_xor(ss, o, 32); }
  float mu  = s * (1.0f / 128.0f);
  float var = ss * (1.0f / 128.0f) - mu * mu;
  float inv = rsqrtf(var + 1e-5f);
  float n[4];
#pragma unroll
  for (int j = 0; j < 4; ++j)
    n[j] = (v[j] - mu) * inv * g[lane * 4 + j] + b[lane * 4 + j];
  float acc[8];
#pragma unroll
  for (int h = 0; h < 8; ++h) {
    float a = 0.f;
#pragma unroll
    for (int j = 0; j < 4; ++j) a += n[j] * W2d[(lane * 4 + j) * 8 + h];
#pragma unroll
    for (int o = 16; o > 0; o >>= 1) a += __shfl_xor(a, o, 32);
    acc[h] = a;
  }
  if (lane == 0) {
    int i = row >> 8, j = row & 255;
#pragma unroll
    for (int h = 0; h < 8; ++h) bias[(h * 256 + i) * 256 + j] = acc[h];
  }
}

// ---------------------------------------------------------------------------
// K3: fused Q/K/V/gate projections: [32768 x 256] x [256 x 256] via WMMA bf16.
// blockIdx.z selects weight. Block = 4 waves; wave w computes a 16x64 strip
// (cols w*64 .. w*64+63) as 4 accumulators sharing one A fragment per K-step.
// z=0 -> q[m][h][i][c] (pre-scaled), z=1 -> k[m][h][j][c],
// z=2 -> vT[m][h][c][j], z=3 -> gate f32 [mi][hc] = sigmoid(. + bgate)
__global__ void k_proj(const bf16* __restrict__ x1n, const bf16* __restrict__ wT,
                       const float* __restrict__ bgate, bf16* __restrict__ q,
                       bf16* __restrict__ kk, bf16* __restrict__ vT,
                       float* __restrict__ gate) {
  int lane  = threadIdx.x & 31;
  int w     = threadIdx.x >> 5;        // 0..3 -> 64-col group
  int z     = blockIdx.z;
  int r0    = blockIdx.x * 16;         // row tile (mi)
  int nbase = w * 64;
  const bf16* WT = wT + z * 65536;

  f32x8 acc[4] = {{}, {}, {}, {}};
  const bf16* arow = x1n + (r0 + (lane & 15)) * 256;
  const bf16* bcol = WT  + (nbase + (lane & 15)) * 256;
#pragma unroll
  for (int k0 = 0; k0 < 256; k0 += 32) {
    bf16x16 af = frag_a(arow + k0, lane);
#pragma unroll
    for (int u = 0; u < 4; ++u) {
      bf16x16 bf_ = frag_b(bcol + u * 16 * 256 + k0, lane);
      acc[u] = WMMA_BF16(af, bf_, acc[u]);
    }
  }
#pragma unroll
  for (int u = 0; u < 4; ++u) {
    int col = nbase + u * 16 + (lane & 15);
    int h = col >> 5, c = col & 31;
#pragma unroll
    for (int v2 = 0; v2 < 8; ++v2) {
      int row = r0 + v2 + ((lane >> 4) << 3);
      int m = row >> 8, i = row & 255;
      float val = acc[u][v2];
      if (z == 0)      q[((m * 8 + h) * 256 + i) * 32 + c] = (bf16)val;
      else if (z == 1) kk[((m * 8 + h) * 256 + i) * 32 + c] = (bf16)val;
      else if (z == 2) vT[((m * 8 + h) * 32 + c) * 256 + i] = (bf16)val;
      else             gate[row * 256 + col] = 1.0f / (1.0f + __expf(-(val + bgate[col])));
    }
  }
}

// ---------------------------------------------------------------------------
// K4: attention per (m,h). Block = 2 waves; each wave owns 8 i-tiles.
// scores staged in LDS (f32), softmax in place (f32x4), P x V via WMMA, gated store.
__global__ void k_attn(const bf16* __restrict__ q, const bf16* __restrict__ kk,
                       const bf16* __restrict__ vT, const float* __restrict__ bias,
                       const float* __restrict__ gate, bf16* __restrict__ attnout) {
  __shared__ float sc[2][16][256];  // 32 KB
  int lane = threadIdx.x & 31;
  int w    = threadIdx.x >> 5;      // 0..1
  int mh   = blockIdx.x;            // 0..1023
  int m = mh >> 3, h = mh & 7;
  const bf16*  qb = q  + mh * 8192;   // [256][32]
  const bf16*  kb = kk + mh * 8192;   // [256][32]
  const bf16*  vb = vT + mh * 8192;   // [32][256]
  const float* bb = bias + h * 65536; // [256][256]
  int koff = (lane >> 4) << 3;

  for (int t = 0; t < 8; ++t) {
    int i0 = (w * 8 + t) * 16;
    // --- scores = q x k^T + bias, staged to LDS ---
    bf16x16 qa = frag_a(qb + (i0 + (lane & 15)) * 32, lane);
    for (int jt = 0; jt < 16; ++jt) {
      int j0 = jt * 16;
      bf16x16 kf = frag_b(kb + (j0 + (lane & 15)) * 32, lane);
      f32x8 acc = {};
      acc = WMMA_BF16(qa, kf, acc);
      int col = j0 + (lane & 15);
#pragma unroll
      for (int v2 = 0; v2 < 8; ++v2) {
        int r = v2 + ((lane >> 4) << 3);
        sc[w][r][col] = acc[v2] + bb[(i0 + r) * 256 + col];
      }
    }
    __syncthreads();
    // --- softmax in place: 2 lanes per row, 128 cols each, f32x4 vectorized ---
    {
      int r = lane >> 1;
      f32x4* rowp = (f32x4*)&sc[w][r][(lane & 1) * 128];
      float mx = -3.0e38f;
      for (int jj = 0; jj < 32; ++jj) {
        f32x4 x = rowp[jj];
        mx = fmaxf(mx, fmaxf(fmaxf(x[0], x[1]), fmaxf(x[2], x[3])));
      }
      mx = fmaxf(mx, __shfl_xor(mx, 1, 32));
      float s = 0.f;
      for (int jj = 0; jj < 32; ++jj) {
        f32x4 x = rowp[jj];
        f32x4 e;
        e[0] = __expf(x[0] - mx); e[1] = __expf(x[1] - mx);
        e[2] = __expf(x[2] - mx); e[3] = __expf(x[3] - mx);
        s += (e[0] + e[1]) + (e[2] + e[3]);
        rowp[jj] = e;
      }
      s += __shfl_xor(s, 1, 32);
      float inv = 1.0f / s;
      for (int jj = 0; jj < 32; ++jj) rowp[jj] = rowp[jj] * inv;
    }
    __syncthreads();
    // --- out(16x32) = P(16x256) x V(256x32) : two 16x16 WMMA accumulators ---
    f32x8 acc0 = {}, acc1 = {};
    const float* pr = &sc[w][lane & 15][0];
#pragma unroll 4
    for (int k0 = 0; k0 < 256; k0 += 32) {
      union { bf16x16 v; bf16x8 hh[2]; } pa;
      f32x4 p0 = *(const f32x4*)(pr + k0 + koff);
      f32x4 p1 = *(const f32x4*)(pr + k0 + koff + 4);
      f32x4 p2 = *(const f32x4*)(pr + k0 + 16 + koff);
      f32x4 p3 = *(const f32x4*)(pr + k0 + 16 + koff + 4);
#pragma unroll
      for (int jj = 0; jj < 4; ++jj) {
        pa.hh[0][jj]     = (bf16)p0[jj];
        pa.hh[0][jj + 4] = (bf16)p1[jj];
        pa.hh[1][jj]     = (bf16)p2[jj];
        pa.hh[1][jj + 4] = (bf16)p3[jj];
      }
      bf16x16 b0 = frag_b(vb + (lane & 15) * 256 + k0, lane);
      bf16x16 b1 = frag_b(vb + (16 + (lane & 15)) * 256 + k0, lane);
      acc0 = WMMA_BF16(pa.v, b0, acc0);
      acc1 = WMMA_BF16(pa.v, b1, acc1);
    }
    // --- gate and store attnout bf16 [mi][hc] ---
#pragma unroll
    for (int v2 = 0; v2 < 8; ++v2) {
      int i = i0 + v2 + ((lane >> 4) << 3);
      int rowoff = (m * 256 + i) * 256;
      int col0 = h * 32 + (lane & 15);
      int col1 = col0 + 16;
      attnout[rowoff + col0] = (bf16)(acc0[v2] * gate[rowoff + col0]);
      attnout[rowoff + col1] = (bf16)(acc1[v2] * gate[rowoff + col1]);
    }
    __syncthreads();
  }
}

// ---------------------------------------------------------------------------
// K5: final projection [32768 x 256] x [256 x 256] + bfinal -> f32 output.
// Block = 4 waves; wave w computes 16x64 strip with 4 accumulators.
__global__ void k_final(const bf16* __restrict__ attn, const bf16* __restrict__ WfT,
                        const float* __restrict__ bfinal, float* __restrict__ out) {
  int lane  = threadIdx.x & 31;
  int w     = threadIdx.x >> 5;  // 0..3
  int r0    = blockIdx.x * 16;
  int nbase = w * 64;
  f32x8 acc[4] = {{}, {}, {}, {}};
  const bf16* arow = attn + (r0 + (lane & 15)) * 256;
  const bf16* bcol = WfT  + (nbase + (lane & 15)) * 256;
#pragma unroll
  for (int k0 = 0; k0 < 256; k0 += 32) {
    bf16x16 af = frag_a(arow + k0, lane);
#pragma unroll
    for (int u = 0; u < 4; ++u) {
      bf16x16 bf_ = frag_b(bcol + u * 16 * 256 + k0, lane);
      acc[u] = WMMA_BF16(af, bf_, acc[u]);
    }
  }
#pragma unroll
  for (int u = 0; u < 4; ++u) {
    int col = nbase + u * 16 + (lane & 15);
    float bf_add = bfinal[col];
#pragma unroll
    for (int v2 = 0; v2 < 8; ++v2) {
      int row = r0 + v2 + ((lane >> 4) << 3);
      out[row * 256 + col] = acc[u][v2] + bf_add;
    }
  }
}

// ---------------------------------------------------------------------------
extern "C" void kernel_launch(void* const* d_in, const int* in_sizes, int n_in,
                              void* d_out, int out_size, void* d_ws, size_t ws_size,
                              hipStream_t stream) {
  const float* x1d    = (const float*)d_in[0];
  const float* x2d    = (const float*)d_in[1];
  const float* ln1_g  = (const float*)d_in[2];
  const float* ln1_b  = (const float*)d_in[3];
  const float* ln2_g  = (const float*)d_in[4];
  const float* ln2_b  = (const float*)d_in[5];
  const float* Wq     = (const float*)d_in[6];
  const float* Wk     = (const float*)d_in[7];
  const float* Wv     = (const float*)d_in[8];
  const float* W2d    = (const float*)d_in[9];
  const float* Wgate  = (const float*)d_in[10];
  const float* bgate  = (const float*)d_in[11];
  const float* Wfinal = (const float*)d_in[12];
  const float* bfinal = (const float*)d_in[13];

  char* ws = (char*)d_ws;
  bf16*  wT    = (bf16*)(ws + 0);            // 5 * 65536 bf16 (WqT,WkT,WvT,WgT | WfT)
  bf16*  x1n   = (bf16*)(ws + 655360);       // 32768*256 bf16
  float* bias  = (float*)(ws + 17432576);    // 8*256*256 f32
  bf16*  qb    = (bf16*)(ws + 19529728);     // [m][h][i][c] bf16
  bf16*  kb    = (bf16*)(ws + 36306944);     // [m][h][j][c] bf16
  bf16*  vTb   = (bf16*)(ws + 53084160);     // [m][h][c][j] bf16
  float* gate  = (float*)(ws + 69861376);    // [mi][hc] f32
  bf16*  attn  = (bf16*)(ws + 103415808);    // [mi][hc] bf16

  k_prep_weights<<<1280, 256, 0, stream>>>(Wq, Wk, Wv, Wgate, Wfinal, wT);
  k_ln1<<<4096, 256, 0, stream>>>(x1d, ln1_g, ln1_b, x1n);
  k_ln2_bias<<<8192, 256, 0, stream>>>(x2d, ln2_g, ln2_b, W2d, bias);
  dim3 g3(2048, 1, 4);
  k_proj<<<g3, 128, 0, stream>>>(x1n, wT, bgate, qb, kb, vTb, gate);
  k_attn<<<1024, 64, 0, stream>>>(qb, kb, vTb, bias, gate, attn);
  k_final<<<2048, 128, 0, stream>>>(attn, wT + 4 * 65536, bfinal, (float*)d_out);
}
